// ConvolutionalSelfAttention_64398739636873
// MI455X (gfx1250) — matrix operand order repaired and checked by reading, not compiled, not run on hardware
//
#include <hip/hip_runtime.h>
#include <hip/hip_bf16.h>

// ---- WMMA vector types (CDNA5 / gfx1250) ----
typedef __attribute__((ext_vector_type(16))) __bf16 v16bf;
typedef __attribute__((ext_vector_type(8)))  float  v8f;

union BFV {
    unsigned short u[16];
    uint4          q[2];   // two aligned 16B halves (halves 0..7 / 8..15)
    v16bf          v;
};

#if defined(__AMDGCN__) && __has_builtin(__builtin_amdgcn_cvt_pk_bf16_f32)
#define HAVE_CVT_PK_BF16 1
#endif

__device__ __forceinline__ unsigned short f2bf_sw(float f) {
    union { float f; unsigned int u; } c; c.f = f;
    unsigned int u = c.u;
    unsigned int r = u + 0x7FFFu + ((u >> 16) & 1u);  // round-to-nearest-even
    return (unsigned short)(r >> 16);
}

// Convert two f32 -> packed bf16x2 in one VALU op when hardware supports it.
__device__ __forceinline__ unsigned int pack2bf(float f0, float f1) {
#ifdef HAVE_CVT_PK_BF16
    auto r = __builtin_amdgcn_cvt_pk_bf16_f32(f0, f1);
    unsigned int u;
    __builtin_memcpy(&u, &r, 4);
    return u;
#else
    return (unsigned)f2bf_sw(f0) | ((unsigned)f2bf_sw(f1) << 16);
#endif
}

__device__ __forceinline__ unsigned short f2bf(float f) {
#ifdef HAVE_CVT_PK_BF16
    return (unsigned short)(pack2bf(f, 0.f) & 0xFFFFu);
#else
    return f2bf_sw(f);
#endif
}

// Problem constants: B=8, C=64, H=W=64, K=7, HC=WC=58, HW=4096
#define XB_ELEMS   2097152   // B*HW*C
#define QKV_STRIDE 2097152   // per-matrix elems

// ---------------------------------------------------------------------------
// Kernel 0a: cast + transpose x [B,C,H,W] f32 -> Xb [B*HW, C] bf16.
// One thread per pixel: 64 coalesced b32 reads, 8 contiguous b128 stores.
// ---------------------------------------------------------------------------
__global__ void csa_cast_x(const float* __restrict__ x,
                           unsigned short* __restrict__ Xb) {
    const int p = blockIdx.x * blockDim.x + threadIdx.x;   // 0..32767 (b*4096+pix)
    if (p >= 32768) return;
    const int b  = p >> 12;
    const int pp = p & 4095;
    const float* src = x + ((long)b << 18) + pp;           // x[b][c][pp], c-stride 4096
    unsigned int buf[32];
#pragma unroll
    for (int cc = 0; cc < 32; ++cc) {
        const float f0 = src[(2 * cc) << 12];
        const float f1 = src[(2 * cc + 1) << 12];
        buf[cc] = pack2bf(f0, f1);
    }
    uint4* dst = (uint4*)(Xb + (long)p * 64);
#pragma unroll
    for (int qq = 0; qq < 8; ++qq) dst[qq] = *(const uint4*)&buf[qq * 4];
}

// ---------------------------------------------------------------------------
// Kernel 0b: cast + transpose weights -> Wt[3][cout][cin] bf16 so the WMMA
// B-operand run (k = hi*16+e over cin) is contiguous.
// ---------------------------------------------------------------------------
__global__ void csa_cast_w(const float* __restrict__ Wq,
                           const float* __restrict__ Wk,
                           const float* __restrict__ Wv,
                           unsigned short* __restrict__ Wt) {
    int idx = blockIdx.x * blockDim.x + threadIdx.x;
    if (idx >= 3 * 4096) return;
    const int m = idx >> 12;
    const int r = idx & 4095;           // r = ci*64 + co
    const int ci = r >> 6, co = r & 63;
    const float* src = (m == 0) ? Wq : ((m == 1) ? Wk : Wv);
    Wt[m * 4096 + co * 64 + ci] = f2bf(src[r]);
}

// ---------------------------------------------------------------------------
// Kernel 1: QKV projection GEMM [32768x64]*[64x64]+bias via WMMA bf16.
// grid=(2048,3), block=32.  Q,K stored [pix][c]; V stored channel-major per
// image row (Vt) so the attention B-operand is contiguous; its D-fragment
// rows land on adjacent Vt columns -> 4 packed b128 stores per lane.
// ---------------------------------------------------------------------------
__global__ void __launch_bounds__(32)
csa_qkv_gemm(const unsigned short* __restrict__ Xb,
             const unsigned short* __restrict__ Wt,
             const float* __restrict__ bq,
             const float* __restrict__ bk,
             const float* __restrict__ bv,
             unsigned short* __restrict__ QKV) {
    const int lane  = threadIdx.x;
    const int m     = lane & 15;
    const int hi    = lane >> 4;
    const int abase = hi ? 8 : 0;
    const int rowBase = blockIdx.x << 4;
    const int mat     = blockIdx.y;

    const unsigned short* Wm   = Wt + mat * 4096;
    const unsigned short* arow = Xb + (long)(rowBase + m) * 64;

    v8f acc[4];
#pragma unroll
    for (int t = 0; t < 4; ++t) acc[t] = v8f{0.f,0.f,0.f,0.f,0.f,0.f,0.f,0.f};

#pragma unroll
    for (int chunk = 0; chunk < 2; ++chunk) {
        const int k0 = chunk * 32;
        BFV a;                                         // A: 16 rows x 32 cin
        a.q[0] = *(const uint4*)(arow + k0 + hi * 16);
        a.q[1] = *(const uint4*)(arow + k0 + hi * 16 + 8);
#pragma unroll
        for (int t = 0; t < 4; ++t) {
            const int col = t * 16 + m;                // B: 32 cin x 16 cout
            const unsigned short* wp = Wm + col * 64 + k0 + hi * 16;
            BFV bm;
            bm.q[0] = *(const uint4*)(wp);
            bm.q[1] = *(const uint4*)(wp + 8);
            acc[t] = __builtin_amdgcn_wmma_f32_16x16x32_bf16(
                false, a.v, false, bm.v, (short)0, acc[t], false, false);
        }
    }

    const float* bias = (mat == 0) ? bq : ((mat == 1) ? bk : bv);
    unsigned short* Out = QKV + (long)mat * QKV_STRIDE;
    const int p0 = rowBase + hi * 8;                   // first D row of this lane half
#pragma unroll
    for (int t = 0; t < 4; ++t) {
        const int col = t * 16 + m;
        const float bb = bias[col];
        if (mat == 2) {
            // Vt layout: [(p>>6)][c][p&63]; rows p0..p0+7 are adjacent columns.
            unsigned int pk[4];
#pragma unroll
            for (int k = 0; k < 4; ++k)
                pk[k] = pack2bf(acc[t][2 * k] + bb, acc[t][2 * k + 1] + bb);
            unsigned short* vp = Out + (((long)(p0 >> 6)) * 64 + col) * 64 + (p0 & 63);
            *(uint4*)vp = *(const uint4*)pk;
        } else {
#pragma unroll
            for (int r = 0; r < 8; ++r)
                Out[(long)(p0 + r) * 64 + col] = f2bf(acc[t][r] + bb);
        }
    }
}

// ---------------------------------------------------------------------------
// Kernel 2: banded local attention, all-WMMA.  grid=(4,58,8), block=32.
// ---------------------------------------------------------------------------
__global__ void __launch_bounds__(32)
csa_attn(const unsigned short* __restrict__ QKV, float* __restrict__ out) {
    const unsigned short* Qb = QKV;
    const unsigned short* Kb = QKV + QKV_STRIDE;
    const unsigned short* Vt = QKV + 2 * QKV_STRIDE;   // [b*64+row][c][col]

    const int lane  = threadIdx.x;
    const int m     = lane & 15;
    const int hi    = lane >> 4;
    const int abase = hi ? 8 : 0;

    const int jbase = blockIdx.x << 4;   // 0,16,32,48
    const int i     = blockIdx.y;        // 0..57
    const int b     = blockIdx.z;        // 0..7

    __shared__ float                      Sco[7][16][32];  // raw scores / exp scratch
    __shared__ alignas(16) unsigned short Aat[7][16][32];  // softmaxed band, bf16

    // ---- Q tile (A operand), both 32-channel chunks in registers ----
    const int cRow = i + 3;
    int cCol = jbase + m + 3; if (cCol > 63) cCol = 63;    // invalid windows clamped
    const unsigned short* qrow = Qb + ((long)((b << 12) + cRow * 64 + cCol)) * 64;
    BFV aQ[2];
#pragma unroll
    for (int chunk = 0; chunk < 2; ++chunk) {
        aQ[chunk].q[0] = *(const uint4*)(qrow + chunk * 32 + hi * 16);
        aQ[chunk].q[1] = *(const uint4*)(qrow + chunk * 32 + hi * 16 + 8);
    }

    // ---- scores via WMMA ----
#pragma unroll
    for (int di = 0; di < 7; ++di) {
        const int kRow = i + di;
        if (di < 6) {  // prefetch next K row (gfx1250 global_prefetch path)
            int pc = jbase + m; if (pc > 63) pc = 63;
            __builtin_prefetch(Kb + ((long)((b << 12) + (kRow + 1) * 64 + pc)) * 64, 0, 1);
        }
        v8f s0 = v8f{0.f,0.f,0.f,0.f,0.f,0.f,0.f,0.f};
        v8f s1 = v8f{0.f,0.f,0.f,0.f,0.f,0.f,0.f,0.f};
#pragma unroll
        for (int chunk = 0; chunk < 2; ++chunk) {
#pragma unroll
            for (int g = 0; g < 2; ++g) {
                int col = jbase + g * 16 + m; if (col > 63) col = 63;  // band cols >=22 unused
                const unsigned short* krow =
                    Kb + ((long)((b << 12) + kRow * 64 + col)) * 64 + chunk * 32 + hi * 16;
                BFV bK;                      // B: 32 channels x 16 key columns
                bK.q[0] = *(const uint4*)(krow);
                bK.q[1] = *(const uint4*)(krow + 8);
                if (g == 0)
                    s0 = __builtin_amdgcn_wmma_f32_16x16x32_bf16(
                        false, aQ[chunk].v, false, bK.v, (short)0, s0, false, false);
                else
                    s1 = __builtin_amdgcn_wmma_f32_16x16x32_bf16(
                        false, aQ[chunk].v, false, bK.v, (short)0, s1, false, false);
            }
        }
#pragma unroll
        for (int r = 0; r < 8; ++r) {
            const int M = r + hi * 8;
            Sco[di][M][m]      = s0[r] * 0.125f;   // 1/sqrt(C)
            Sco[di][M][16 + m] = s1[r] * 0.125f;
        }
    }
    __syncthreads();

    // ---- zero the banded-attn A matrix (vectorized) ----
    {
        uint4* p = (uint4*)&Aat[0][0][0];            // 7*16*32*2B = 448 uint4
        const uint4 z = {0u, 0u, 0u, 0u};
        for (int t = lane; t < 448; t += 32) p[t] = z;
    }
    __syncthreads();

    // ---- softmax over the 49 band entries, one window per lane 0..15 ----
    if (hi == 0) {
        const int j = m;
        float mx = -1e30f;
#pragma unroll
        for (int di = 0; di < 7; ++di)
#pragma unroll
            for (int dj = 0; dj < 7; ++dj)
                mx = fmaxf(mx, Sco[di][j][j + dj]);
        float sum = 0.f;
#pragma unroll
        for (int di = 0; di < 7; ++di)
#pragma unroll
            for (int dj = 0; dj < 7; ++dj) {
                const float e = __expf(Sco[di][j][j + dj] - mx);
                Sco[di][j][j + dj] = e;
                sum += e;
            }
        const float inv = 1.0f / sum;
#pragma unroll
        for (int di = 0; di < 7; ++di)
#pragma unroll
            for (int dj = 0; dj < 7; ++dj)
                Aat[di][j][j + dj] = f2bf(Sco[di][j][j + dj] * inv);
    }
    __syncthreads();

    // ---- output: O = band-attn @ V via WMMA (B operand straight from Vt) ----
    v8f o[4];
#pragma unroll
    for (int t = 0; t < 4; ++t) o[t] = v8f{0.f,0.f,0.f,0.f,0.f,0.f,0.f,0.f};

#pragma unroll
    for (int di = 0; di < 7; ++di) {
        BFV aA;   // A: 16 windows x 32 key positions (banded bf16 from LDS)
        aA.q[0] = *(const uint4*)&Aat[di][m][hi * 16];
        aA.q[1] = *(const uint4*)&Aat[di][m][hi * 16 + 8];
        const long slab = (long)((b << 6) + i + di);   // image row index
#pragma unroll
        for (int t = 0; t < 4; ++t) {
            const int c = t * 16 + m;
            const unsigned short* vp = Vt + (slab * 64 + c) * 64 + jbase + hi * 16;
            BFV bV;   // B: 32 key positions x 16 channels (contiguous in col!)
            bV.q[0] = *(const uint4*)(vp);
            bV.q[1] = *(const uint4*)(vp + 8);
            o[t] = __builtin_amdgcn_wmma_f32_16x16x32_bf16(
                false, aA.v, false, bV.v, (short)0, o[t], false, false);
        }
    }

    // ---- store [B, 58, 58, 64] f32 ----
#pragma unroll
    for (int t = 0; t < 4; ++t) {
#pragma unroll
        for (int r = 0; r < 8; ++r) {
            const int j  = r + hi * 8;
            const int jj = jbase + j;
            if (jj < 58)
                out[((((long)b * 58) + i) * 58 + jj) * 64 + t * 16 + m] = o[t][r];
        }
    }
}

// ---------------------------------------------------------------------------
extern "C" void kernel_launch(void* const* d_in, const int* in_sizes, int n_in,
                              void* d_out, int out_size, void* d_ws, size_t ws_size,
                              hipStream_t stream) {
    const float* x  = (const float*)d_in[0];
    const float* Wq = (const float*)d_in[1];
    const float* bq = (const float*)d_in[2];
    const float* Wk = (const float*)d_in[3];
    const float* bk = (const float*)d_in[4];
    const float* Wv = (const float*)d_in[5];
    const float* bv = (const float*)d_in[6];
    float* out = (float*)d_out;

    // workspace layout (bf16 stored as u16):
    //   Xb  [B*HW, 64]                    : 2,097,152 elems (4 MiB)
    //   QKV [3] regions of 2,097,152      : Q,K row-major; V channel-major/row
    //   pad 64 elems (tail overreads of Vt)
    //   Wt  [3][64][64] transposed weights: 12,288 elems
    unsigned short* Xb  = (unsigned short*)d_ws;
    unsigned short* QKV = Xb + XB_ELEMS;
    unsigned short* Wt  = QKV + 3 * QKV_STRIDE + 64;

    csa_cast_x<<<128, 256, 0, stream>>>(x, Xb);
    csa_cast_w<<<(3 * 4096 + 255) / 256, 256, 0, stream>>>(Wq, Wk, Wv, Wt);
    csa_qkv_gemm<<<dim3(2048, 3, 1), 32, 0, stream>>>(Xb, Wt, bq, bk, bv, QKV);
    csa_attn<<<dim3(4, 58, 8), 32, 0, stream>>>(QKV, out);
}
